// Crosscoder_13280038879396
// MI455X (gfx1250) — matrix operand order repaired and checked by reading, not compiled, not run on hardware
//
#include <hip/hip_runtime.h>
#include <stdint.h>

// ---------------- problem constants ----------------
#define BB   1024
#define LL   4
#define DD   768
#define FF   16384
#define KIN  3072            // L*D
#define NZP  (BB * FF)       // 16,777,216 elements in z / z_pre

// output layout (floats): x_hat | z | z_pre | loss | per_layer_mse
#define OFF_XHAT 0
#define OFF_Z    3145728
#define OFF_ZPRE 19922944
#define OFF_LOSS 36700160
#define OFF_PLM  36700161

// workspace layout (bytes)
#define WS_HIST    0          // 4096 u32
#define WS_STATE   16384      // 16 u32
#define WS_ROWCNT  16448      // 1024 u32
#define WS_ROWLIST 20544      // 1024 * 256 * uint2  = 2 MB
#define WS_PARTIAL 2117696    // 1024*4 floats

#define ROW_CAP 256

typedef __attribute__((ext_vector_type(16))) __bf16 v16bf;
typedef __attribute__((ext_vector_type(8)))  float  v8f;
typedef __attribute__((ext_vector_type(4)))  int    v4i;

union Frag { uint32_t u[8]; v16bf h; };
union Acc  { float    f[8]; v8f   v; };

__device__ __forceinline__ uint32_t f2bf(float a) {
    uint32_t u = __float_as_uint(a);
    return (u + 0x7FFFu + ((u >> 16) & 1u)) >> 16;   // round-to-nearest-even
}

// ---- CDNA5 async global->LDS path (guarded; falls back to sync copy) ----
#if __has_builtin(__builtin_amdgcn_global_load_async_to_lds_b128) && \
    __has_builtin(__builtin_amdgcn_s_wait_asynccnt)
#define HAVE_ASYNC_LDS 1
__device__ __forceinline__ void async_cp16(const void* g, void* l) {
    __builtin_amdgcn_global_load_async_to_lds_b128(
        (__attribute__((address_space(1))) v4i*)(g),
        (__attribute__((address_space(3))) v4i*)(l), 0, 0);
}
#else
#define HAVE_ASYNC_LDS 0
#endif

// ---------------- init workspace ----------------
__global__ void init_ws(uint32_t* hist, uint32_t* state, uint32_t* row_cnt, const int* kptr) {
    int i = blockIdx.x * blockDim.x + threadIdx.x;
    if (i < 4096) hist[i] = 0;
    if (i < 1024) row_cnt[i] = 0;
    if (i == 0) {
        state[0] = 0;                                  // prefix bits
        state[1] = (uint32_t)(BB * (*kptr));           // k remaining (total_k)
        state[2] = 0;                                  // tie counter
        state[3] = 0;                                  // final threshold bits
        state[4] = 0;                                  // ties to keep
    }
}

// ---------------- xc = bf16(x - b_pre), packed pairs ----------------
__global__ __launch_bounds__(256) void prep_xc(const float* __restrict__ x,
                                               const float* __restrict__ b_pre,
                                               uint32_t* __restrict__ xc) {
    int i = blockIdx.x * blockDim.x + threadIdx.x;    // over B * KIN/2
    if (i >= BB * (KIN / 2)) return;
    int jp = i % (KIN / 2);
    int b  = i / (KIN / 2);
    float v0 = x[(size_t)b * KIN + 2 * jp]     - b_pre[2 * jp];
    float v1 = x[(size_t)b * KIN + 2 * jp + 1] - b_pre[2 * jp + 1];
    xc[i] = f2bf(v0) | (f2bf(v1) << 16);
}

// ---------------- encoder GEMM: z_pre = xc @ W_enc + b_enc ----------------
// 128x128 tile / workgroup (8 waves, each 64x32 = 4x2 WMMA tiles), K-chunk 32,
// double-buffered LDS; A via async global->LDS, B pipelined through registers.
#define LDSS  20            // LDS row stride in dwords (16B-aligned, bank-spread)
#define BUFSZ (128 * LDSS)  // dwords per buffer
#define NCHUNK (KIN / 32)   // 96

__global__ __launch_bounds__(256) void encoder_gemm(const uint32_t* __restrict__ xc,
                                                    const float*    __restrict__ W_enc,
                                                    const float*    __restrict__ b_enc,
                                                    float*          __restrict__ zpre) {
    __shared__ __align__(16) uint32_t ldsA[2 * BUFSZ];
    __shared__ __align__(16) uint32_t ldsB[2 * BUFSZ];

    const int tid  = threadIdx.x;
    const int lane = tid & 31;
    const int wid  = tid >> 5;
    const int wm   = wid >> 2;          // 0..1  (64-row slice)
    const int wn   = wid & 3;           // 0..3  (32-col slice)
    const int m0   = blockIdx.y * 128;
    const int n0   = blockIdx.x * 128;

    const int row16 = lane & 15;
    const int halfA = (lane >> 4) << 2; // 0 or 4  (A: K-pair offset per lane half)
    const int koffB = (lane >> 4) << 2; // 0 or 4  (B: K-pair offset per lane half)

    // B staging map: 2 items/thread, item = (kp 0..15, nq 0..31)
    const int nq0 = tid & 31,         kp0 = tid >> 5;
    const int nq1 = nq0,              kp1 = kp0 + 8;          // idx+256 => kp += 8
    // A staging map: 2 items/thread, item = (m 0..127, q 0..3)
    const int mA0 = tid >> 2,         qA0 = tid & 3;
    const int mA1 = (tid + 256) >> 2, qA1 = qA0;

    Acc c[4][2];
#pragma unroll
    for (int i = 0; i < 4; ++i)
#pragma unroll
        for (int j = 0; j < 2; ++j)
#pragma unroll
            for (int v = 0; v < 8; ++v) c[i][j].f[v] = 0.0f;

    float4 rb0a, rb0b, rb1a, rb1b;      // B f32 pipeline registers
    {   // prologue: B regs for chunk 0
        const float* p0 = W_enc + (size_t)(2 * kp0) * FF + n0 + 4 * nq0;
        const float* p1 = W_enc + (size_t)(2 * kp1) * FF + n0 + 4 * nq1;
        rb0a = *(const float4*)p0;  rb0b = *(const float4*)(p0 + FF);
        rb1a = *(const float4*)p1;  rb1b = *(const float4*)(p1 + FF);
    }
#if HAVE_ASYNC_LDS
    // prologue: async A for chunk 0 into buffer 0
    async_cp16(xc + (size_t)(m0 + mA0) * (KIN / 2) + (qA0 << 2), &ldsA[mA0 * LDSS + (qA0 << 2)]);
    async_cp16(xc + (size_t)(m0 + mA1) * (KIN / 2) + (qA1 << 2), &ldsA[mA1 * LDSS + (qA1 << 2)]);
#endif

    for (int it = 0; it < NCHUNK; ++it) {
        const int kc  = it * 32;
        const int cur = it & 1;
        uint32_t* lA = ldsA + cur * BUFSZ;
        uint32_t* lB = ldsB + cur * BUFSZ;

        // ---- store B (f32 regs -> packed bf16 pairs, [n][kpair] layout) ----
        lB[(4 * nq0 + 0) * LDSS + kp0] = f2bf(rb0a.x) | (f2bf(rb0b.x) << 16);
        lB[(4 * nq0 + 1) * LDSS + kp0] = f2bf(rb0a.y) | (f2bf(rb0b.y) << 16);
        lB[(4 * nq0 + 2) * LDSS + kp0] = f2bf(rb0a.z) | (f2bf(rb0b.z) << 16);
        lB[(4 * nq0 + 3) * LDSS + kp0] = f2bf(rb0a.w) | (f2bf(rb0b.w) << 16);
        lB[(4 * nq1 + 0) * LDSS + kp1] = f2bf(rb1a.x) | (f2bf(rb1b.x) << 16);
        lB[(4 * nq1 + 1) * LDSS + kp1] = f2bf(rb1a.y) | (f2bf(rb1b.y) << 16);
        lB[(4 * nq1 + 2) * LDSS + kp1] = f2bf(rb1a.z) | (f2bf(rb1b.z) << 16);
        lB[(4 * nq1 + 3) * LDSS + kp1] = f2bf(rb1a.w) | (f2bf(rb1b.w) << 16);

#if !HAVE_ASYNC_LDS
        // ---- sync A copy (fallback): b128 load + b128 LDS store ----
        {
            const uint4 v0 = *(const uint4*)(xc + (size_t)(m0 + mA0) * (KIN / 2) + (kc >> 1) + (qA0 << 2));
            const uint4 v1 = *(const uint4*)(xc + (size_t)(m0 + mA1) * (KIN / 2) + (kc >> 1) + (qA1 << 2));
            *(uint4*)&lA[mA0 * LDSS + (qA0 << 2)] = v0;
            *(uint4*)&lA[mA1 * LDSS + (qA1 << 2)] = v1;
        }
#endif
        // ---- launch next chunk's B global loads (hidden behind compute) ----
        if (it + 1 < NCHUNK) {
            const int kn = kc + 32;
            const float* p0 = W_enc + (size_t)(kn + 2 * kp0) * FF + n0 + 4 * nq0;
            const float* p1 = W_enc + (size_t)(kn + 2 * kp1) * FF + n0 + 4 * nq1;
            rb0a = *(const float4*)p0;  rb0b = *(const float4*)(p0 + FF);
            rb1a = *(const float4*)p1;  rb1b = *(const float4*)(p1 + FF);
        }

#if HAVE_ASYNC_LDS
        __builtin_amdgcn_s_wait_asynccnt(0);   // this chunk's A tiles landed in LDS
#endif
        __syncthreads();

#if HAVE_ASYNC_LDS
        // safe past the barrier: every wave finished reading the other buffer
        if (it + 1 < NCHUNK) {
            uint32_t* lAn = ldsA + (1 - cur) * BUFSZ;
            const int kn2 = (kc + 32) >> 1;
            async_cp16(xc + (size_t)(m0 + mA0) * (KIN / 2) + kn2 + (qA0 << 2),
                       &lAn[mA0 * LDSS + (qA0 << 2)]);
            async_cp16(xc + (size_t)(m0 + mA1) * (KIN / 2) + kn2 + (qA1 << 2),
                       &lAn[mA1 * LDSS + (qA1 << 2)]);
        }
#endif
        // ---- fragments + 8 WMMA ----
        Frag a[4], bfr[2];
#pragma unroll
        for (int i = 0; i < 4; ++i) {
            int row = wm * 64 + i * 16 + row16;
            *(uint4*)&a[i].u[0] = *(const uint4*)&lA[row * LDSS + halfA];
            *(uint4*)&a[i].u[4] = *(const uint4*)&lA[row * LDSS + 8 + halfA];
        }
#pragma unroll
        for (int j = 0; j < 2; ++j) {
            int col = wn * 32 + j * 16 + row16;
            *(uint4*)&bfr[j].u[0] = *(const uint4*)&lB[col * LDSS + koffB];
            *(uint4*)&bfr[j].u[4] = *(const uint4*)&lB[col * LDSS + koffB + 8];
        }
#pragma unroll
        for (int i = 0; i < 4; ++i)
#pragma unroll
            for (int j = 0; j < 2; ++j)
                c[i][j].v = __builtin_amdgcn_wmma_f32_16x16x32_bf16(
                    false, a[i].h, false, bfr[j].h, (short)0, c[i][j].v, false, false);
    }

    // epilogue: C layout — lane<16: M=v, lane>=16: M=v+8; N = lane&15
    const int rbase = m0 + wm * 64 + ((lane >> 4) << 3);
#pragma unroll
    for (int i = 0; i < 4; ++i) {
#pragma unroll
        for (int j = 0; j < 2; ++j) {
            int col = n0 + wn * 32 + j * 16 + row16;
            float be = b_enc[col];
#pragma unroll
            for (int v = 0; v < 8; ++v) {
                int m = rbase + i * 16 + v;
                zpre[(size_t)m * FF + col] = c[i][j].f[v] + be;
            }
        }
    }
}

// ---------------- radix-select histogram over positive-float bit patterns ----
__global__ __launch_bounds__(256) void hist_pass(const float* __restrict__ zpre,
                                                 uint32_t* __restrict__ hist,
                                                 const uint32_t* __restrict__ state,
                                                 int shift, uint32_t binmask,
                                                 uint32_t prefix_mask) {
    __shared__ uint32_t h[4096];
    for (int i = threadIdx.x; i < 4096; i += blockDim.x) h[i] = 0;
    __syncthreads();
    const uint32_t prefix = state[0] & prefix_mask;
    for (size_t i = (size_t)blockIdx.x * blockDim.x + threadIdx.x; i < (size_t)NZP;
         i += (size_t)gridDim.x * blockDim.x) {
        float v = zpre[i];
        if (v > 0.0f) {
            uint32_t bits = __float_as_uint(v);
            if ((bits & prefix_mask) == prefix)
                atomicAdd(&h[(bits >> shift) & binmask], 1u);
        }
    }
    __syncthreads();
    for (int i = threadIdx.x; i < 4096; i += blockDim.x)
        if (h[i]) atomicAdd(&hist[i], h[i]);
}

__global__ void hist_select(uint32_t* hist, uint32_t* state, int shift, int nbins,
                            int is_final) {
    if (threadIdx.x == 0) {
        uint32_t krem = state[1];
        uint32_t prefix = state[0];
        int sel = 0;
        for (int bin = nbins - 1; bin >= 0; --bin) {
            uint32_t cnt = hist[bin];
            if (cnt >= krem) { sel = bin; break; }
            krem -= cnt;
        }
        prefix |= ((uint32_t)sel) << shift;
        state[0] = prefix;
        state[1] = krem;
        if (is_final) { state[3] = prefix; state[4] = krem; }
    }
    __syncthreads();
    for (int i = threadIdx.x; i < 4096; i += blockDim.x) hist[i] = 0;
}

// ---------------- z = topk-masked relu(z_pre) ----------------
__global__ __launch_bounds__(256) void z_write(const float* __restrict__ zpre,
                                               float* __restrict__ z,
                                               uint32_t* state) {
    const uint32_t T = state[3];
    const uint32_t keep_eq = state[4];
    for (size_t i = (size_t)blockIdx.x * blockDim.x + threadIdx.x; i < (size_t)NZP;
         i += (size_t)gridDim.x * blockDim.x) {
        float v = zpre[i];
        float o = 0.0f;
        if (v > 0.0f) {
            uint32_t bits = __float_as_uint(v);
            if (bits > T) o = v;
            else if (bits == T) { if (atomicAdd(&state[2], 1u) < keep_eq) o = v; }
        }
        z[i] = o;
    }
}

// ---------------- deterministic per-row nonzero compaction (1 wave / row) ----
__global__ __launch_bounds__(32) void row_compact(const float* __restrict__ z,
                                                  uint32_t* __restrict__ row_cnt,
                                                  uint2* __restrict__ rowlist) {
    const int b = blockIdx.x;
    const int lane = threadIdx.x;
    uint32_t cnt = 0;
    for (int base = 0; base < FF; base += 32) {
        float v = z[(size_t)b * FF + base + lane];
        bool p = (v != 0.0f);
        uint32_t m = (uint32_t)__ballot(p);
        uint32_t pre = __popc(m & ((1u << lane) - 1u));
        if (p) {
            uint32_t slot = cnt + pre;
            if (slot < ROW_CAP)
                rowlist[(size_t)b * ROW_CAP + slot] = make_uint2((uint32_t)(base + lane),
                                                                __float_as_uint(v));
        }
        cnt += __popc(m);
    }
    if (lane == 0) row_cnt[b] = cnt < ROW_CAP ? cnt : ROW_CAP;
}

// ---------------- sparse decoder: x_hat = z @ W_dec + b_dec ----------------
__global__ __launch_bounds__(256) void decoder(const uint32_t* __restrict__ row_cnt,
                                               const uint2* __restrict__ rowlist,
                                               const float* __restrict__ W_dec,
                                               const float* __restrict__ b_dec,
                                               float* __restrict__ x_hat) {
    const int b = blockIdx.x, l = blockIdx.y, t = threadIdx.x;
    float a0 = b_dec[l * DD + t];
    float a1 = b_dec[l * DD + 256 + t];
    float a2 = b_dec[l * DD + 512 + t];
    const uint32_t cnt = row_cnt[b];
    const float* Wl = W_dec + (size_t)l * FF * DD;
    const uint2* lst = rowlist + (size_t)b * ROW_CAP;
    for (uint32_t s = 0; s < cnt; ++s) {
        uint2 e = lst[s];
        float v = __uint_as_float(e.y);
        const float* w = Wl + (size_t)e.x * DD;
        a0 += v * w[t];
        a1 += v * w[256 + t];
        a2 += v * w[512 + t];
    }
    float* o = x_hat + (size_t)b * (LL * DD) + l * DD;
    o[t] = a0; o[256 + t] = a1; o[512 + t] = a2;
}

// ---------------- deterministic MSE reduction ----------------
__global__ __launch_bounds__(256) void mse_partial(const float* __restrict__ x,
                                                   const float* __restrict__ x_hat,
                                                   float* __restrict__ partial) {
    const int b = blockIdx.x, t = threadIdx.x;
    __shared__ float red[256];
    for (int l = 0; l < LL; ++l) {
        const float* xr = x + (size_t)b * (LL * DD) + l * DD;
        const float* hr = x_hat + (size_t)b * (LL * DD) + l * DD;
        float s = 0.0f;
        for (int r = 0; r < 3; ++r) {
            float d = xr[t + 256 * r] - hr[t + 256 * r];
            s += d * d;
        }
        red[t] = s;
        __syncthreads();
        for (int off = 128; off > 0; off >>= 1) {
            if (t < off) red[t] += red[t + off];
            __syncthreads();
        }
        if (t == 0) partial[b * LL + l] = red[0];
        __syncthreads();
    }
}

__global__ void mse_final(const float* __restrict__ partial, float* loss, float* plm) {
    const int t = threadIdx.x;
    __shared__ float pl[LL];
    if (t < LL) {
        float s = 0.0f;
        for (int b = 0; b < BB; ++b) s += partial[b * LL + t];
        s /= (float)(BB * DD);
        plm[t] = s;
        pl[t] = s;
    }
    __syncthreads();
    if (t == 0) loss[0] = pl[0] + pl[1] + pl[2] + pl[3];
}

// ---------------- launcher ----------------
extern "C" void kernel_launch(void* const* d_in, const int* in_sizes, int n_in,
                              void* d_out, int out_size, void* d_ws, size_t ws_size,
                              hipStream_t stream) {
    const float* x     = (const float*)d_in[0];
    const float* b_pre = (const float*)d_in[1];
    const float* W_enc = (const float*)d_in[2];
    const float* b_enc = (const float*)d_in[3];
    const float* W_dec = (const float*)d_in[4];
    const float* b_dec = (const float*)d_in[5];
    const int*   kptr  = (const int*)d_in[6];

    float* out   = (float*)d_out;
    float* x_hat = out + OFF_XHAT;
    float* z     = out + OFF_Z;
    float* zpre  = out + OFF_ZPRE;
    float* loss  = out + OFF_LOSS;
    float* plm   = out + OFF_PLM;

    uint8_t*  ws      = (uint8_t*)d_ws;
    uint32_t* hist    = (uint32_t*)(ws + WS_HIST);
    uint32_t* state   = (uint32_t*)(ws + WS_STATE);
    uint32_t* row_cnt = (uint32_t*)(ws + WS_ROWCNT);
    uint2*    rowlist = (uint2*)(ws + WS_ROWLIST);
    float*    partial = (float*)(ws + WS_PARTIAL);

    // xc (bf16-packed activations) parked in the z output region (overwritten later)
    uint32_t* xc = (uint32_t*)z;

    init_ws<<<16, 256, 0, stream>>>(hist, state, row_cnt, kptr);
    prep_xc<<<(BB * (KIN / 2) + 255) / 256, 256, 0, stream>>>(x, b_pre, xc);
    encoder_gemm<<<dim3(FF / 128, BB / 128), 256, 0, stream>>>(xc, W_enc, b_enc, zpre);

    hist_pass<<<512, 256, 0, stream>>>(zpre, hist, state, 19, 4095u, 0u);
    hist_select<<<1, 256, 0, stream>>>(hist, state, 19, 4096, 0);
    hist_pass<<<512, 256, 0, stream>>>(zpre, hist, state, 7, 4095u, 0x7FF80000u);
    hist_select<<<1, 256, 0, stream>>>(hist, state, 7, 4096, 0);
    hist_pass<<<512, 256, 0, stream>>>(zpre, hist, state, 0, 127u, 0x7FFFFF80u);
    hist_select<<<1, 256, 0, stream>>>(hist, state, 0, 128, 1);

    z_write<<<2048, 256, 0, stream>>>(zpre, z, state);
    row_compact<<<BB, 32, 0, stream>>>(z, row_cnt, rowlist);
    decoder<<<dim3(BB, LL), 256, 0, stream>>>(row_cnt, rowlist, W_dec, b_dec, x_hat);
    mse_partial<<<BB, 256, 0, stream>>>(x, x_hat, partial);
    mse_final<<<1, 64, 0, stream>>>(partial, loss, plm);
}
